// CausalAttention_64871186038836
// MI455X (gfx1250) — matrix operand order repaired and checked by reading, not compiled, not run on hardware
//
#include <hip/hip_runtime.h>
#include <hip/hip_bf16.h>

// ---------------------------------------------------------------------------
// out = softmax((xWq)(xWk)^T / sqrt(d)) (xWv)   [no mask, per reference]
// B=2, S=4096, D=512, fp32 in/out. bf16 WMMA, fp32 accumulation.
// v4: Q-tile staged with GLOBAL_LOAD_ASYNC_TO_LDS_B128 (ASYNCcnt path).
// ---------------------------------------------------------------------------

constexpr int BATCH  = 2;
constexpr int SEQ    = 4096;
constexpr int DMODEL = 512;
constexpr int MTOT   = BATCH * SEQ;

typedef __attribute__((ext_vector_type(16))) __bf16        v16bf;
typedef __attribute__((ext_vector_type(8)))  unsigned int  v8u32;
typedef __attribute__((ext_vector_type(8)))  float         v8f;
typedef __attribute__((ext_vector_type(4)))  unsigned int  u32x4;

union Frag { u32x4 q[2]; v8u32 w; };

// round-to-nearest (ties away) bf16: bias then keep high 16 bits
__device__ __forceinline__ unsigned short f32_to_bf16(float f) {
  return (unsigned short)((__builtin_bit_cast(unsigned int, f) + 0x8000u) >> 16);
}
// two floats -> packed bf16x2 with a single v_perm_b32 (bytes {b3,b2,a3,a2})
__device__ __forceinline__ unsigned int pack_bf16x2(float a, float b) {
  const unsigned int ua = __builtin_bit_cast(unsigned int, a) + 0x8000u;
  const unsigned int ub = __builtin_bit_cast(unsigned int, b) + 0x8000u;
  return __builtin_amdgcn_perm(ub, ua, 0x07060302u);
}

__device__ __forceinline__ v8f wmma_bf16(v8u32 a, v8u32 b, v8f c) {
  return __builtin_amdgcn_wmma_f32_16x16x32_bf16(
      false, __builtin_bit_cast(v16bf, a),
      false, __builtin_bit_cast(v16bf, b),
      (short)0, c, false, false);
}

// A-fragment (16x32 bf16): dwords 0-3 @ ushort off hi*16, dwords 4-7 @ hi*16+8
__device__ __forceinline__ Frag load_afrag(const unsigned short* row, int hi) {
  Frag a;
  a.q[0] = *(const u32x4*)(row + hi * 16);
  a.q[1] = *(const u32x4*)(row + hi * 16 + 8);
  return a;
}
// B-fragment (32x16 bf16): dword j = K-pair (hi*16+2j); contiguous 32 bytes
__device__ __forceinline__ Frag load_bfrag(const unsigned short* base) {
  Frag b;
  b.q[0] = *(const u32x4*)(base);
  b.q[1] = *(const u32x4*)(base + 8);
  return b;
}

// ---------------------------------------------------------------------------
// Kernel 1: q/k = x@W row-major bf16; v = x@Wv stored TRANSPOSED vt[b][d][s].
// block 256 = 8 waves; tile 128 rows x 64 cols; x/W^T staged in LDS as bf16.
// grid = (MTOT/128, DMODEL/64, 3)
// ---------------------------------------------------------------------------
__global__ __launch_bounds__(256) void qkv_proj_kernel(
    const float* __restrict__ x,
    const float* __restrict__ Wq, const float* __restrict__ Wk,
    const float* __restrict__ Wv,
    unsigned short* __restrict__ qb, unsigned short* __restrict__ kb,
    unsigned short* __restrict__ vt)
{
  __shared__ unsigned short sX[128 * 32];   // 8 KB  x-tile  [row][k] bf16
  __shared__ unsigned short sWt[64 * 32];   // 4 KB  W-tile  [n][k]  bf16

  const int lane = threadIdx.x & 31, wave = threadIdx.x >> 5;
  const int l = lane & 15, hi = lane >> 4;
  const int m0 = blockIdx.x * 128;
  const int n0 = blockIdx.y * 64;
  const float* W = (blockIdx.z == 0) ? Wq : ((blockIdx.z == 1) ? Wk : Wv);

  const int sx_row  = threadIdx.x >> 1;
  const int sx_half = threadIdx.x & 1;
  const int sw_n    = threadIdx.x & 63;
  const int sw_kg   = threadIdx.x >> 6;

  v8f acc[4] = {};

  for (int k0 = 0; k0 < DMODEL; k0 += 32) {
    { // stage x tile (converted once, reused by 8 waves)
      const float* src = x + (size_t)(m0 + sx_row) * DMODEL + k0 + sx_half * 16;
      unsigned int pk[8];
#pragma unroll
      for (int i = 0; i < 8; ++i) pk[i] = pack_bf16x2(src[2 * i], src[2 * i + 1]);
      u32x4* dst = (u32x4*)(sX + sx_row * 32 + sx_half * 16);
      dst[0] = (u32x4){pk[0], pk[1], pk[2], pk[3]};
      dst[1] = (u32x4){pk[4], pk[5], pk[6], pk[7]};
    }
    { // stage W tile transposed: sWt[n][k] -> B K-pairs contiguous
      unsigned int pk[4];
#pragma unroll
      for (int i = 0; i < 4; ++i) {
        const size_t r0 = (size_t)(k0 + sw_kg * 8 + 2 * i) * DMODEL + n0 + sw_n;
        pk[i] = pack_bf16x2(W[r0], W[r0 + DMODEL]);
      }
      *(u32x4*)(sWt + sw_n * 32 + sw_kg * 8) = (u32x4){pk[0], pk[1], pk[2], pk[3]};
    }
    __syncthreads();

    const Frag a = load_afrag(sX + (wave * 16 + l) * 32, hi);
#pragma unroll
    for (int n = 0; n < 4; ++n) {
      const Frag b = load_bfrag(sWt + (n * 16 + l) * 32 + hi * 16);
      acc[n] = wmma_bf16(a.w, b.w, acc[n]);
    }
    __syncthreads();
  }

  if (blockIdx.z == 2) {               // V: store transposed vt[b][d][s]
#pragma unroll
    for (int n = 0; n < 4; ++n)
#pragma unroll
      for (int r = 0; r < 8; ++r) {
        const int m   = m0 + hi * 8 + r;
        const int bat = m >> 12;
        const int s   = m & (SEQ - 1);
        const int col = n0 + n * 16 + l;
        vt[((size_t)bat * DMODEL + col) * SEQ + s] = f32_to_bf16(acc[n][r]);
      }
  } else {
    unsigned short* out = (blockIdx.z == 0) ? qb : kb;
#pragma unroll
    for (int n = 0; n < 4; ++n)
#pragma unroll
      for (int r = 0; r < 8; ++r)
        out[(size_t)(m0 + hi * 8 + r) * DMODEL + n0 + n * 16 + l] =
            f32_to_bf16(acc[n][r]);
  }
}

// ---------------------------------------------------------------------------
// Kernel 2: flash attention. Workgroup = 32 queries (2 subtiles/wave),
// 8 waves, key tile 128. Every K/V fragment load is shared by 2 WMMAs.
// grid = (SEQ/32, BATCH)
// ---------------------------------------------------------------------------
__global__ __launch_bounds__(256) void attention_kernel(
    const unsigned short* __restrict__ qb, const unsigned short* __restrict__ kb,
    const unsigned short* __restrict__ vt, float* __restrict__ out)
{
  const int lane = threadIdx.x & 31, wave = threadIdx.x >> 5;
  const int l = lane & 15, hi = lane >> 4;
  const int batch = blockIdx.y;
  const int q0    = blockIdx.x * 32;

  const unsigned short* Q  = qb + (size_t)batch * SEQ * DMODEL;
  const unsigned short* K  = kb + (size_t)batch * SEQ * DMODEL;
  const unsigned short* Vt = vt + (size_t)batch * DMODEL * SEQ;

  __shared__ unsigned short sQ[32 * DMODEL];   // 32 KB
  __shared__ float          sS[32 * 128];      // 16 KB
  __shared__ unsigned short sP[32 * 128];      //  8 KB
  __shared__ float          sRed[32 * 8];      //  1 KB
  __shared__ float          sMax[32], sSum[32], sCorr[32];

  { // stage 32-query tile via async global->LDS DMA (ASYNCcnt-tracked)
    const unsigned int lds_base = (unsigned int)(size_t)(void*)sQ;
    const unsigned short* gsrc = Q + (size_t)q0 * DMODEL;
    for (int i = threadIdx.x; i < 32 * DMODEL / 8; i += 256) {
      const unsigned long long ga =
          (unsigned long long)(size_t)(gsrc + (size_t)i * 8);
      const unsigned int la = lds_base + (unsigned int)i * 16u;
      asm volatile("global_load_async_to_lds_b128 %0, %1, off"
                   :: "v"(la), "v"(ga) : "memory");
    }
    asm volatile("s_wait_asynccnt 0x0" ::: "memory");
  }
  if (threadIdx.x < 32) { sMax[threadIdx.x] = -1e30f; sSum[threadIdx.x] = 0.0f; }
  __syncthreads();

  v8f acc[2][4] = {};                           // 2 q-subtiles x 64 d-cols
  const float scale = 0.044194173824159216f;    // 1/sqrt(512)

  for (int kt = 0; kt < SEQ; kt += 128) {
    if (kt + 128 < SEQ) {
      __builtin_prefetch(K  + (size_t)(kt + 128 + wave * 16) * DMODEL, 0, 0);
      __builtin_prefetch(Vt + (size_t)(wave * 64) * SEQ + kt + 128, 0, 0);
    }

    // ---- phase 1: S[32q x 16k], keys kt+16*wave; B frag shared by 2 qt ----
    const int keyb = kt + wave * 16;
    v8f sacc[2] = {};
    for (int k0 = 0; k0 < DMODEL; k0 += 32) {
      const Frag b  = load_bfrag(K + (size_t)(keyb + l) * DMODEL + k0 + hi * 16);
      const Frag a0 = load_afrag(sQ + l * DMODEL + k0, hi);
      const Frag a1 = load_afrag(sQ + (16 + l) * DMODEL + k0, hi);
      sacc[0] = wmma_bf16(a0.w, b.w, sacc[0]);
      sacc[1] = wmma_bf16(a1.w, b.w, sacc[1]);
    }
#pragma unroll
    for (int qt = 0; qt < 2; ++qt)
#pragma unroll
      for (int r = 0; r < 8; ++r)
        sS[(qt * 16 + hi * 8 + r) * 128 + wave * 16 + l] = sacc[qt][r] * scale;
    __syncthreads();

    // ---- phase 2: cooperative online softmax (32 rows x 128 cols) ----
    const int row = threadIdx.x >> 3;           // 8 threads per row
    const int t8  = threadIdx.x & 7;            // 16 cols each
    float vals[16], lm = -1e30f;
#pragma unroll
    for (int c = 0; c < 16; ++c) {
      vals[c] = sS[row * 128 + t8 * 16 + c];
      lm = fmaxf(lm, vals[c]);
    }
    sRed[row * 8 + t8] = lm;
    __syncthreads();
    if (t8 == 0) {
      float m = sMax[row];
#pragma unroll
      for (int c = 0; c < 8; ++c) m = fmaxf(m, sRed[row * 8 + c]);
      sCorr[row] = __expf(sMax[row] - m);
      sMax[row]  = m;
    }
    __syncthreads();
    const float mnew = sMax[row];
    float ls = 0.0f;
#pragma unroll
    for (int c = 0; c < 8; ++c) {               // pack P as bf16 pairs
      const float p0 = __expf(vals[2 * c] - mnew);
      const float p1 = __expf(vals[2 * c + 1] - mnew);
      ls += p0 + p1;
      ((unsigned int*)sP)[(row * 128 + t8 * 16) / 2 + c] = pack_bf16x2(p0, p1);
    }
    sRed[row * 8 + t8] = ls;
    __syncthreads();
    if (t8 == 0) {
      float s = 0.0f;
#pragma unroll
      for (int c = 0; c < 8; ++c) s += sRed[row * 8 + c];
      sSum[row] = sSum[row] * sCorr[row] + s;
    }

    // rescale running output by softmax correction
#pragma unroll
    for (int qt = 0; qt < 2; ++qt) {
#pragma unroll
      for (int r = 0; r < 8; ++r) {
        const float c = sCorr[qt * 16 + hi * 8 + r];
#pragma unroll
        for (int n = 0; n < 4; ++n) acc[qt][n][r] *= c;
      }
    }

    // ---- phase 3: acc += P[32x128] @ V[128 x 64-slice]; B shared by 2 qt --
#pragma unroll
    for (int kk = 0; kk < 128; kk += 32) {
      const Frag a0 = load_afrag(sP + l * 128 + kk, hi);
      const Frag a1 = load_afrag(sP + (16 + l) * 128 + kk, hi);
#pragma unroll
      for (int n = 0; n < 4; ++n) {
        const int dcol = wave * 64 + n * 16 + l;
        const Frag b = load_bfrag(Vt + (size_t)dcol * SEQ + kt + kk + hi * 16);
        acc[0][n] = wmma_bf16(a0.w, b.w, acc[0][n]);
        acc[1][n] = wmma_bf16(a1.w, b.w, acc[1][n]);
      }
    }
    __syncthreads();
  }

  // epilogue: normalize with v_rcp, store fp32
#pragma unroll
  for (int qt = 0; qt < 2; ++qt) {
    float rinv[8];
#pragma unroll
    for (int r = 0; r < 8; ++r)
      rinv[r] = __builtin_amdgcn_rcpf(sSum[qt * 16 + hi * 8 + r]);
#pragma unroll
    for (int n = 0; n < 4; ++n)
#pragma unroll
      for (int r = 0; r < 8; ++r) {
        const int row = q0 + qt * 16 + hi * 8 + r;
        const int col = wave * 64 + n * 16 + l;
        out[((size_t)batch * SEQ + row) * DMODEL + col] = acc[qt][n][r] * rinv[r];
      }
  }
}

// ---------------------------------------------------------------------------
extern "C" void kernel_launch(void* const* d_in, const int* in_sizes, int n_in,
                              void* d_out, int out_size, void* d_ws, size_t ws_size,
                              hipStream_t stream) {
  const float* x  = (const float*)d_in[0];
  const float* Wq = (const float*)d_in[1];
  const float* Wk = (const float*)d_in[2];
  const float* Wv = (const float*)d_in[3];
  float* out = (float*)d_out;

  unsigned short* qb = (unsigned short*)d_ws;                 // [b][s][d] bf16
  unsigned short* kb = qb + (size_t)BATCH * SEQ * DMODEL;     // [b][s][d] bf16
  unsigned short* vt = kb + (size_t)BATCH * SEQ * DMODEL;     // [b][d][s] bf16

  dim3 gp(MTOT / 128, DMODEL / 64, 3);
  qkv_proj_kernel<<<gp, 256, 0, stream>>>(x, Wq, Wk, Wv, qb, kb, vt);

  dim3 ga(SEQ / 32, BATCH);
  attention_kernel<<<ga, 256, 0, stream>>>(qb, kb, vt, out);
}
